// ResonanceLayer_28432683499853
// MI455X (gfx1250) — matrix operand order repaired and checked by reading, not compiled, not run on hardware
//
#include <hip/hip_runtime.h>
#include <math.h>

// ---------------- problem constants ----------------
#define NS     32768     // n_samples
#define WW     2048      // window
#define CD     32        // control dim == n_res
#define NR     32        // resonances
#define XP     4         // expressivity
#define NO     16        // oscillators
#define FR     32        // frames
#define ENVC   256       // env coeffs
#define STEP   1024      // NS / FR

// ---------------- workspace layout (float indices into d_ws) ----------------
#define P_PARAMS 0         // 2048 osc * 4 floats  {coef, xc, omega, phi}
#define P_NORM2  8192      // 128
#define P_DSM    8320      // 4*32 softmaxed deformation weights
#define P_MIXW   8448      // 32*2 dry/wet softmax
#define P_ABSG   8512      // 32 |gains|
#define P_ROUTED 8576      // 32*32 routed[r][f]
#define P_IMP    9600      // 32*2048 impulse f32
#define P_BASE   75136     // 32*4*32768 base = imp (*) res, f32
#define RESH_BYTE_OFF 17077760ull   // (P_BASE + 4194304)*4 ; f16 normalized res, 8MB
#define IMPH_BYTE_OFF 25466368ull   // RESH + 8MB ; f16 impulse, 128KB (TDM source)

typedef __attribute__((ext_vector_type(16))) _Float16 v16h;
typedef __attribute__((ext_vector_type(8)))  float    v8f;
typedef __attribute__((ext_vector_type(2)))  float    v2f;
typedef __attribute__((ext_vector_type(4)))  unsigned int v4u;
typedef __attribute__((ext_vector_type(8)))  unsigned int v8u;

union HV32 { uint4 q[2]; v16h v; };

__device__ __forceinline__ float sigm(float x) { return 1.0f / (1.0f + expf(-x)); }

// ============================================================================
// K0: per-oscillator params, deformation softmax, mix softmax, |gains|
// ============================================================================
__global__ void prep_kernel(const float* __restrict__ mass,
                            const float* __restrict__ damping,
                            const float* __restrict__ tension,
                            const float* __restrict__ idisp,
                            const float* __restrict__ amps,
                            const float* __restrict__ deform,
                            const float* __restrict__ mix,
                            const float* __restrict__ gains,
                            float* __restrict__ Wks) {
  int tid = threadIdx.x;
  // oscillator params: flat i = o*128 + r*4 + x
  for (int i = tid; i < NO * NR * XP; i += blockDim.x) {
    float m   = sigm(mass[i]);
    float dmp = sigm(damping[i]) * 30.0f;
    float ten = exp10f(tension[i]);
    float idp = idisp[i];
    float xc  = dmp / (2.0f * m);
    float om  = sqrtf(fmaxf(ten - xc * xc, 1e-12f));
    float phi = atan2f(xc * idp, idp * om);
    float amp = idp / cosf(phi);
    float a2  = amps[i] * amps[i];
    Wks[P_PARAMS + 4 * i + 0] = amp * a2;
    Wks[P_PARAMS + 4 * i + 1] = xc;
    Wks[P_PARAMS + 4 * i + 2] = om;
    Wks[P_PARAMS + 4 * i + 3] = phi;
  }
  if (tid < FR) {  // deformation softmax over X per frame
    float v[XP], mx = -1e30f;
    for (int x = 0; x < XP; ++x) { v[x] = (x == 0 ? 1.0f : 0.0f) + deform[x * FR + tid]; mx = fmaxf(mx, v[x]); }
    float s = 0.0f;
    for (int x = 0; x < XP; ++x) { v[x] = expf(v[x] - mx); s += v[x]; }
    for (int x = 0; x < XP; ++x) Wks[P_DSM + x * FR + tid] = v[x] / s;
  }
  if (tid < NR) {  // dry/wet softmax + |gains|
    float e0 = expf(mix[2 * tid]), e1 = expf(mix[2 * tid + 1]);
    float s = e0 + e1;
    Wks[P_MIXW + 2 * tid]     = e0 / s;
    Wks[P_MIXW + 2 * tid + 1] = e1 / s;
    Wks[P_ABSG + tid] = fabsf(gains[tid]);
  }
}

// ============================================================================
// K1: routed[r,f] = sum_c control[c,f]*router[c,r]  via V_WMMA_F32_16X16X4_F32
// ============================================================================
__global__ void routing_wmma_kernel(const float* __restrict__ control,
                                    const float* __restrict__ router,
                                    float* __restrict__ out,
                                    float* __restrict__ Wks) {
  int lane = threadIdx.x & 31;
  int wv   = threadIdx.x >> 5;
  int f0 = (wv & 1) * 16;       // M tile (frames)
  int r0 = (wv >> 1) * 16;      // N tile (resonances)
  int mn = lane & 15;
  bool hi = lane >= 16;
  v8f c = {};
  for (int k0 = 0; k0 < CD; k0 += 4) {
    v2f a, b;
    int ka = k0 + (hi ? 2 : 0);
    int kb = k0 + (hi ? 3 : 1);
    a.x = control[ka * FR + f0 + mn];
    a.y = control[kb * FR + f0 + mn];
    b.x = router[ka * NR + r0 + mn];
    b.y = router[kb * NR + r0 + mn];
    c = __builtin_amdgcn_wmma_f32_16x16x4_f32(false, a, false, b, (short)0, c, false, false);
  }
#pragma unroll
  for (int i = 0; i < 8; ++i) {
    int f  = f0 + (hi ? 8 : 0) + i;   // C row M
    int rr = r0 + mn;                 // C col N
    float v = c[i];
    out[NS + rr * FR + f] = v;        // before_upsample, (R,F) row-major
    Wks[P_ROUTED + rr * FR + f] = v;
  }
}

// ============================================================================
// K2a: sum of squares of ramped resonance per (r,x)   (one block per rx)
// K2b: recompute, normalize, store f16
// ============================================================================
__global__ void res_norm_kernel(float* __restrict__ Wks) {
  __shared__ float sp[NO * 4];
  __shared__ float red[256];
  int rx = blockIdx.x, tid = threadIdx.x;
  if (tid < NO * 4) { int o = tid >> 2, j = tid & 3; sp[tid] = Wks[P_PARAMS + 4 * (o * 128 + rx) + j]; }
  __syncthreads();
  const float DT = 10.0f / 32767.0f;
  float acc = 0.0f;
  for (int t = tid; t < NS; t += 256) {
    float tt = t * DT, s = 0.0f;
#pragma unroll
    for (int o = 0; o < NO; ++o)
      s += sp[4 * o] * expf(-sp[4 * o + 1] * tt) * cosf(sp[4 * o + 2] * tt - sp[4 * o + 3]);
    if (t < 10) s *= t * (1.0f / 9.0f);
    acc += s * s;
  }
  red[tid] = acc; __syncthreads();
  for (int s = 128; s > 0; s >>= 1) { if (tid < s) red[tid] += red[tid + s]; __syncthreads(); }
  if (tid == 0) Wks[P_NORM2 + rx] = red[0];
}

__global__ void res_store_kernel(float* __restrict__ Wks, _Float16* __restrict__ resh) {
  __shared__ float sp[NO * 4];
  int rx = blockIdx.x, tid = threadIdx.x;
  if (tid < NO * 4) { int o = tid >> 2, j = tid & 3; sp[tid] = Wks[P_PARAMS + 4 * (o * 128 + rx) + j]; }
  __syncthreads();
  float scale = 1.0f / (sqrtf(Wks[P_NORM2 + rx]) + 1e-8f);
  const float DT = 10.0f / 32767.0f;
  for (int t = tid; t < NS; t += 256) {
    float tt = t * DT, s = 0.0f;
#pragma unroll
    for (int o = 0; o < NO; ++o)
      s += sp[4 * o] * expf(-sp[4 * o + 1] * tt) * cosf(sp[4 * o + 2] * tt - sp[4 * o + 3]);
    if (t < 10) s *= t * (1.0f / 9.0f);
    resh[rx * NS + t] = (_Float16)(s * scale);
  }
}

// ============================================================================
// K3: impulse = interp(attack,W)^2 * noise  (f32 for final mix, f16 for TDM)
// ============================================================================
__global__ void imp_kernel(const float* __restrict__ attack,
                           const float* __restrict__ noise,
                           float* __restrict__ Wks,
                           _Float16* __restrict__ imph) {
  int g = blockIdx.x * blockDim.x + threadIdx.x;   // 32*2048
  int c = g >> 11, j = g & (WW - 1);
  float pos = fminf(fmaxf((j + 0.5f) * 0.125f - 0.5f, 0.0f), (float)(ENVC - 1));
  int lo = (int)floorf(pos);
  int hi = min(lo + 1, ENVC - 1);
  float w = pos - (float)lo;
  float e = attack[c * ENVC + lo] * (1.0f - w) + attack[c * ENVC + hi] * w;
  float v = e * e * noise[c * WW + j];
  Wks[P_IMP + g] = v;
  imph[g] = (_Float16)v;
}

// ============================================================================
// K4: base[r,x,u] = sum_m imp[r,u-m]*res[r,x,m]  — Toeplitz GEMM on WMMA f16.
//     imp tile staged global->LDS by the Tensor Data Mover (TENSORcnt),
//     A = imp Toeplitz gathered from LDS, B = res rows, C accum f32.
// ============================================================================
__global__ void __launch_bounds__(128) conv_wmma_kernel(const float* __restrict__ Wks,
                                                        const _Float16* __restrict__ resh,
                                                        const _Float16* __restrict__ imph,
                                                        float* __restrict__ base) {
  __shared__ _Float16 impl[WW];
  int r = blockIdx.y;

  // ---- TDM: async-copy 2048 f16 (4KB) of imp[r] into LDS, wave 0 only ----
  if (threadIdx.x < 32) {
    unsigned long long ga = (unsigned long long)(uintptr_t)(imph + r * WW);
    unsigned lds_addr = (unsigned)(uintptr_t)&impl[0];   // flat addr[31:0] == LDS offset
    v4u g0;
    g0[0] = 1u;                                           // count=1 valid descriptor
    g0[1] = lds_addr;                                     // D#.lds_addr (bytes)
    g0[2] = (unsigned)(ga & 0xFFFFFFFFu);                 // global_addr[31:0]
    g0[3] = (unsigned)((ga >> 32) & 0x01FFFFFFu) | (2u << 30);  // addr[56:32] | type=2
    v8u g1;
    g1[0] = 1u << 16;                  // wg_mask=0 (not in cluster), data_size=1 (2B)
    g1[1] = ((unsigned)WW) << 16;      // tensor_dim0[15:0] @ bits63:48  (=2048)
    g1[2] = 1u << 16;                  // tensor_dim0 hi=0 | tensor_dim1=1 lo16
    g1[3] = ((unsigned)WW) << 16;      // tensor_dim1 hi=0 | tile_dim0=2048
    g1[4] = 1u;                        // tile_dim1=1, tile_dim2=0
    g1[5] = (unsigned)WW;              // tensor_dim0_stride[31:0]
    g1[6] = 0u;                        // stride hi | dim1_stride lo
    g1[7] = 0u;
    asm volatile("tensor_load_to_lds %0, %1" :: "s"(g0), "s"(g1) : "memory");
    __builtin_amdgcn_s_wait_tensorcnt(0);
  }
  __syncthreads();

  int wv   = threadIdx.x >> 5;
  int lane = threadIdx.x & 31;
  int utile = blockIdx.x * 4 + wv;
  int u0 = utile << 4;
  int uu = lane & 15;
  bool hi = lane >= 16;
  int n = lane & 15;                      // N column (x when < 4)
  const _Float16* brow = resh + ((r * XP + (n < XP ? n : 0)) << 15);

  int mt0 = (u0 >= WW - 1) ? ((u0 - (WW - 1)) >> 5) : 0;
  int mt1 = (u0 + 15) >> 5;
  v8f acc = {};
  for (int mt = mt0; mt <= mt1; ++mt) {
    int m0 = mt << 5;
    int c0 = u0 - m0;
    // --- B fragment: 32 consecutive f16 of res (16 per half-wave) ---
    HV32 hv;
    if (n < XP) {
      const uint4* p = reinterpret_cast<const uint4*>(brow + m0 + (hi ? 16 : 0));
      hv.q[0] = p[0]; hv.q[1] = p[1];
      if (mt + 1 <= mt1) __builtin_prefetch((const void*)(p + 2), 0, 1);
    } else {
      hv.q[0] = make_uint4(0, 0, 0, 0); hv.q[1] = make_uint4(0, 0, 0, 0);
    }
    // --- A fragment: Toeplitz gather from LDS imp ---
    v16h a;
    if (c0 >= 31 && c0 <= WW - 16) {          // interior tile: no range guard
#pragma unroll
      for (int i = 0; i < 16; ++i) {
        int kk = (hi ? 16 : 0) + i;
        a[i] = impl[c0 + uu - kk];
      }
      acc = __builtin_amdgcn_wmma_f32_16x16x32_f16(false, a, false, hv.v, (short)0, acc, false, false);
    } else {                                   // boundary tile: guarded gather
#pragma unroll
      for (int i = 0; i < 16; ++i) {
        int kk = (hi ? 16 : 0) + i;
        int dd = c0 + uu - kk;
        a[i] = ((unsigned)dd < (unsigned)WW) ? impl[dd] : (_Float16)0.0f;
      }
      acc = __builtin_amdgcn_wmma_f32_16x16x32_f16(false, a, false, hv.v, (short)0, acc, false, false);
    }
  }
  if (n < XP) {
    float* outp = base + P_BASE + ((r * XP + n) << 15) + u0 + (hi ? 8 : 0);
#pragma unroll
    for (int i = 0; i < 8; ++i) outp[i] = acc[i];
  }
}

// ============================================================================
// K5: dry/wet mix, deformation interp, tanh, sum over resonances
// ============================================================================
__global__ void final_kernel(const float* __restrict__ Wks, float* __restrict__ out) {
  __shared__ float s_routed[NR * FR];
  __shared__ float s_mixw[2 * NR];
  __shared__ float s_absg[NR];
  __shared__ float s_dsm[XP * FR];
  int tid = threadIdx.x;
  for (int i = tid; i < NR * FR; i += 256) s_routed[i] = Wks[P_ROUTED + i];
  if (tid < 2 * NR)  s_mixw[tid] = Wks[P_MIXW + tid];
  if (tid < NR)      s_absg[tid] = Wks[P_ABSG + tid];
  if (tid < XP * FR) s_dsm[tid]  = Wks[P_DSM + tid];
  __syncthreads();

  int t = blockIdx.x * 256 + tid;
  // deformation weights interpolated at t
  float pos = fminf(fmaxf((t + 0.5f) * (32.0f / 32768.0f) - 0.5f, 0.0f), 31.0f);
  int lo = (int)floorf(pos);
  int hif = min(lo + 1, 31);
  float w = pos - (float)lo;
  float d[XP];
#pragma unroll
  for (int x = 0; x < XP; ++x)
    d[x] = s_dsm[x * FR + lo] * (1.0f - w) + s_dsm[x * FR + hif] * w;

  int kmax = t >> 10;
  int i0 = t - (kmax << 10);
  float accum = 0.0f;
  for (int r = 0; r < NR; ++r) {
    // dry: routed_c[r,t] (<=2 comb taps hit the 2048-wide impulse)
    float dry = s_routed[r * FR + kmax] * Wks[P_IMP + r * WW + i0];
    if (kmax >= 1) dry += s_routed[r * FR + kmax - 1] * Wks[P_IMP + r * WW + i0 + STEP];
    // wet: sum_x d[x] * sum_k routed[r,k] * base[r,x,t-1024k]
    float wet = 0.0f;
#pragma unroll
    for (int x = 0; x < XP; ++x) {
      const float* bp = Wks + P_BASE + ((r * XP + x) << 15);
      float s = 0.0f;
      for (int k = 0; k <= kmax; ++k) s += s_routed[r * FR + k] * bp[t - (k << 10)];
      wet += d[x] * s;
    }
    float v = s_mixw[2 * r] * dry + s_mixw[2 * r + 1] * wet;
    accum += tanhf(v * s_absg[r]);
  }
  out[t] = accum;
}

// ============================================================================
extern "C" void kernel_launch(void* const* d_in, const int* in_sizes, int n_in,
                              void* d_out, int out_size, void* d_ws, size_t ws_size,
                              hipStream_t stream) {
  const float* control = (const float*)d_in[0];
  const float* deform  = (const float*)d_in[1];
  const float* mass    = (const float*)d_in[2];
  const float* damping = (const float*)d_in[3];
  const float* tension = (const float*)d_in[4];
  const float* idisp   = (const float*)d_in[5];
  const float* amps    = (const float*)d_in[6];
  const float* attack  = (const float*)d_in[7];
  const float* router  = (const float*)d_in[8];
  const float* mix     = (const float*)d_in[9];
  const float* gains   = (const float*)d_in[10];
  const float* noise   = (const float*)d_in[11];
  float* out = (float*)d_out;
  float* Wks = (float*)d_ws;
  _Float16* resh = (_Float16*)((char*)d_ws + RESH_BYTE_OFF);
  _Float16* imph = (_Float16*)((char*)d_ws + IMPH_BYTE_OFF);

  prep_kernel<<<1, 128, 0, stream>>>(mass, damping, tension, idisp, amps,
                                     deform, mix, gains, Wks);
  routing_wmma_kernel<<<1, 128, 0, stream>>>(control, router, out, Wks);
  res_norm_kernel<<<128, 256, 0, stream>>>(Wks);
  res_store_kernel<<<128, 256, 0, stream>>>(Wks, resh);
  imp_kernel<<<256, 256, 0, stream>>>(attack, noise, Wks, imph);
  conv_wmma_kernel<<<dim3(512, 32), 128, 0, stream>>>(Wks, resh, imph, Wks);
  final_kernel<<<128, 256, 0, stream>>>(Wks, out);
}
